// SubPixelMotionRefinement_81088982548615
// MI455X (gfx1250) — matrix operand order repaired and checked by reading, not compiled, not run on hardware
//
#include <hip/hip_runtime.h>

// ---------------------------------------------------------------------------
// SubPixelMotionRefinement for MI455X (gfx1250, wave32, WMMA).
// 3x3 convs: block-cooperative implicit GEMM on v_wmma_f32_16x16x32_f16 with
// activations staged into LDS via global_load_async_to_lds_b128 (ASYNCcnt).
// Activations: channel-last f16, channels padded to {32,64,96,128}.
// Weights: pre-packed into A-fragment VGPR order with contiguous k (k = 16*(lane/16) + e).
// ---------------------------------------------------------------------------

typedef __attribute__((ext_vector_type(16))) _Float16 v16h;
typedef __attribute__((ext_vector_type(8)))  float    v8f;

#define HH   256
#define WWD  448
#define HWP  (HH * WWD)      // 114688 pixels per image
#define BBN  2
#define NPIX (BBN * HWP)     // 229376
#define PSTRIP 64            // pixels per block strip (448 = 7*64)

union FragU { uint4 u[2]; v16h h; };   // 32B = one WMMA A/B operand

static __device__ __forceinline__ v8f wmma_f16(const FragU& a, const FragU& b, v8f c) {
  return __builtin_amdgcn_wmma_f32_16x16x32_f16(false, a.h, false, b.h,
                                                (short)0, c, false, false);
}

// --------------------------- input pad: (B,3,H,W) f32 -> (P,32) f16 --------
__global__ __launch_bounds__(256) void pad_input_kernel(
    const float* __restrict__ in, _Float16* __restrict__ out) {
  int pix = blockIdx.x * blockDim.x + threadIdx.x;
  if (pix >= NPIX) return;
  int b = pix / HWP, rem = pix % HWP;
  union { uint4 u[4]; _Float16 h[32]; } o;
  #pragma unroll
  for (int i = 0; i < 32; ++i) o.h[i] = (_Float16)0.f;
  #pragma unroll
  for (int ci = 0; ci < 3; ++ci)
    o.h[ci] = (_Float16)in[((long)b * 3 + ci) * HWP + rem];
  uint4* p = (uint4*)(out + (long)pix * 32);
  p[0] = o.u[0]; p[1] = o.u[1]; p[2] = o.u[2]; p[3] = o.u[3];
}

// ------------------- weight pack: OIHW f32 -> A-fragment-ordered f16 -------
// Packed layout: [mtile][kt][lane][16 halves]; A-operand mapping used here (contiguous k, matches LDS B reads):
//   m = lane%16, h = lane/16, element e:  k = h*16 + e
__global__ __launch_bounds__(256) void pack_weight_kernel(
    const float* __restrict__ w, _Float16* __restrict__ wp,
    int CIN, int CINP, int COUT, int MT, int KT) {
  int t = blockIdx.x * blockDim.x + threadIdx.x;   // one per (mtile,kt,lane)
  if (t >= MT * KT * 32) return;
  int lane = t & 31;
  int kt    = (t >> 5) % KT;
  int mtile = (t >> 5) / KT;
  int m = mtile * 16 + (lane & 15);
  int hh = lane >> 4;
  union { uint4 u[2]; _Float16 h[16]; } o;
  #pragma unroll
  for (int e = 0; e < 16; ++e) {
    int v = e >> 1, s = e & 1;
    int klocal = hh * 16 + e;
    int kg  = kt * 32 + klocal;
    int tap = kg / CINP;            // 0..8  -> (ky,kx)
    int ci  = kg % CINP;
    float val = 0.f;
    if (m < COUT && ci < CIN)
      val = w[((long)m * CIN + ci) * 9 + tap];
    o.h[e] = (_Float16)val;
  }
  uint4* p = (uint4*)(wp + (long)t * 16);
  p[0] = o.u[0]; p[1] = o.u[1];
}

// --------------------------- 3x3 conv via WMMA + LDS staging ---------------
// Block = one 64-pixel row strip; waves = 2*(COUTP/16): wave w -> m-tile w%MT,
// pixel half w/MT.  LDS holds 3 halo rows x 66 pixels x CINP f16, staged with
// async-to-LDS copies (zero border written with ds stores).
template <int CINP>
__global__ void conv3x3_wmma_lds_kernel(
    const _Float16* __restrict__ act, const _Float16* __restrict__ wpack,
    const float* __restrict__ bias, _Float16* __restrict__ out,
    float* __restrict__ out_final, const float* __restrict__ flow,
    int MT, int COUTP, int relu, int final_mode) {
  constexpr int KT   = CINP * 9 / 32;   // 32-K chunks
  constexpr int CCH  = CINP / 16;       // 32B chunks per pixel
  extern __shared__ _Float16 smem[];    // 3 * 66 * CINP halves

  const int tid  = threadIdx.x;
  const int nthr = blockDim.x;          // 64 * MT
  const int pixS = blockIdx.x * PSTRIP;
  const int b    = pixS / HWP;
  const int rem  = pixS - b * HWP;
  const int y    = rem / WWD;
  const int x0   = rem - y * WWD;

  // ---- stage halo rows y-1..y+1, x0-1..x0+64 (zero-padded) into LDS ------
  const int NCHUNK = 3 * 66 * CCH;
  const uint32_t lds_base = (uint32_t)(uintptr_t)smem;
  for (int c = tid; c < NCHUNK; c += nthr) {
    int r    = c / (66 * CCH);
    int rest = c - r * (66 * CCH);
    int p    = rest / CCH;
    int cc   = rest - p * CCH;
    int yy = y + r - 1;
    int xx = x0 + p - 1;
    if (yy >= 0 && yy < HH && xx >= 0 && xx < WWD) {
      const _Float16* g = act + (((long)b * HH + yy) * WWD + xx) * CINP + cc * 16;
      uint32_t loff = lds_base + (uint32_t)c * 32u;
      asm volatile("global_load_async_to_lds_b128 %0, %1, off"
                   :: "v"(loff), "v"(g) : "memory");
    } else {
      *(uint4*)(smem + (size_t)c * 16) = make_uint4(0u, 0u, 0u, 0u);
    }
  }
  asm volatile("s_wait_asynccnt 0x0" ::: "memory");
  __syncthreads();

  // ---- per-wave 16(M) x 32(N) implicit GEMM over K = CINP*9 ---------------
  const int wave = tid >> 5;
  const int lane = tid & 31;
  const int mi = wave % MT;
  const int ni = wave / MT;             // 0 or 1 (32-pixel half of strip)
  const int n  = lane & 15;
  const int hh = lane >> 4;

  v8f c0 = {}; v8f c1 = {};
  const uint4* wp4 = (const uint4*)wpack;
  long wbase = ((long)mi * KT * 32 + lane) * 2;   // uint4 index, +=2 per kt

  for (int tap = 0; tap < 9; ++tap) {
    int dyt = tap / 3, dxt = tap % 3;
    // LDS row pointer for this lane's pixel (+dx) and K-half
    const _Float16* rowp =
        smem + ((size_t)dyt * 66 + (ni * 32 + n + dxt)) * CINP + hh * 16;
    #pragma unroll
    for (int cc = 0; cc < CINP / 32; ++cc) {
      FragU a, b0, b1;
      a.u[0] = wp4[wbase]; a.u[1] = wp4[wbase + 1]; wbase += 2;
      const uint4* p0 = (const uint4*)(rowp + cc * 32);
      const uint4* p1 = (const uint4*)(rowp + 16 * CINP + cc * 32);
      b0.u[0] = p0[0]; b0.u[1] = p0[1];
      b1.u[0] = p1[0]; b1.u[1] = p1[1];
      c0 = wmma_f16(a, b0, c0);
      c1 = wmma_f16(a, b1, c1);
    }
  }

  // ---- epilogue -----------------------------------------------------------
  const int mb = mi * 16 + hh * 8;      // 8 consecutive out-channels
  const int pix0 = pixS + ni * 32;
  if (final_mode) {
    if (hh == 0) {                      // COUT=2 head: flow residual, f32 NCHW
      int xA = x0 + ni * 32 + n, xB = xA + 16;
      long o0 = ((long)b * 2 + 0) * HWP + (long)y * WWD;
      long o1 = ((long)b * 2 + 1) * HWP + (long)y * WWD;
      out_final[o0 + xA] = flow[o0 + xA] + c0[0] + bias[0];
      out_final[o1 + xA] = flow[o1 + xA] + c0[1] + bias[1];
      out_final[o0 + xB] = flow[o0 + xB] + c1[0] + bias[0];
      out_final[o1 + xB] = flow[o1 + xB] + c1[1] + bias[1];
    }
  } else {
    const float4* bp = (const float4*)(bias + mb);
    float4 bb0 = bp[0], bb1 = bp[1];
    float ba[8] = {bb0.x, bb0.y, bb0.z, bb0.w, bb1.x, bb1.y, bb1.z, bb1.w};
    union { uint4 u; _Float16 h[8]; } o0, o1;
    #pragma unroll
    for (int r = 0; r < 8; ++r) {
      float v0 = c0[r] + ba[r], v1 = c1[r] + ba[r];
      if (relu) { v0 = v0 > 0.f ? v0 : 0.f; v1 = v1 > 0.f ? v1 : 0.f; }
      o0.h[r] = (_Float16)v0; o1.h[r] = (_Float16)v1;
    }
    *(uint4*)(out + (long)(pix0 + n)      * COUTP + mb) = o0.u;
    *(uint4*)(out + (long)(pix0 + 16 + n) * COUTP + mb) = o1.u;
  }
}

// ------------------ bilinear warp (border clamp), channel-last f16 ---------
__global__ __launch_bounds__(256) void warp_kernel(
    const _Float16* __restrict__ f2, const float* __restrict__ flow,
    _Float16* __restrict__ out) {
  int pix = blockIdx.x * blockDim.x + threadIdx.x;
  if (pix >= NPIX) return;
  int b = pix / HWP, rem = pix % HWP, y = rem / WWD, x = rem % WWD;
  float fx = (float)x + flow[((long)b * 2 + 0) * HWP + rem];
  float fy = (float)y + flow[((long)b * 2 + 1) * HWP + rem];
  fx = fminf(fmaxf(fx, 0.f), (float)(WWD - 1));
  fy = fminf(fmaxf(fy, 0.f), (float)(HH - 1));
  float x0f = floorf(fx), y0f = floorf(fy);
  int x0 = (int)x0f, y0 = (int)y0f;
  int x1 = min(x0 + 1, WWD - 1), y1 = min(y0 + 1, HH - 1);
  float wx = fx - x0f, wy = fy - y0f;
  const _Float16* p00 = f2 + (((long)b * HH + y0) * WWD + x0) * 32;
  const _Float16* p01 = f2 + (((long)b * HH + y0) * WWD + x1) * 32;
  const _Float16* p10 = f2 + (((long)b * HH + y1) * WWD + x0) * 32;
  const _Float16* p11 = f2 + (((long)b * HH + y1) * WWD + x1) * 32;
  _Float16* po = out + (long)pix * 32;
  #pragma unroll
  for (int ci = 0; ci < 32; ++ci) {
    float v = ((float)p00[ci] * (1.f - wx) + (float)p01[ci] * wx) * (1.f - wy) +
              ((float)p10[ci] * (1.f - wx) + (float)p11[ci] * wx) * wy;
    po[ci] = (_Float16)v;
  }
}

// -------- cost volume: 81 integer offsets (clamped lookup) + flow, C=96 ----
__global__ __launch_bounds__(256) void costvol_kernel(
    const _Float16* __restrict__ f1, const _Float16* __restrict__ f2w,
    const float* __restrict__ flow, _Float16* __restrict__ cv) {
  int pix = blockIdx.x * blockDim.x + threadIdx.x;
  if (pix >= NPIX) return;
  int b = pix / HWP, rem = pix % HWP, y = rem / WWD, x = rem % WWD;
  float a[32];
  const _Float16* pf1 = f1 + (long)pix * 32;
  #pragma unroll
  for (int ci = 0; ci < 32; ++ci) a[ci] = (float)pf1[ci];
  _Float16* po = cv + (long)pix * 96;
  for (int o = 0; o < 81; ++o) {             // o = (dy+4)*9 + (dx+4)
    int dy = o / 9 - 4, dx = o % 9 - 4;
    int yy = min(max(y + dy, 0), HH - 1);
    int xx = min(max(x + dx, 0), WWD - 1);
    const _Float16* pg = f2w + (((long)b * HH + yy) * WWD + xx) * 32;
    float acc = 0.f;
    #pragma unroll
    for (int ci = 0; ci < 32; ++ci) acc += a[ci] * (float)pg[ci];
    po[o] = (_Float16)acc;
  }
  po[81] = (_Float16)flow[((long)b * 2 + 0) * HWP + rem];
  po[82] = (_Float16)flow[((long)b * 2 + 1) * HWP + rem];
  #pragma unroll
  for (int c2 = 83; c2 < 96; ++c2) po[c2] = (_Float16)0.f;
}

// ---------------------------------------------------------------------------
extern "C" void kernel_launch(void* const* d_in, const int* in_sizes, int n_in,
                              void* d_out, int out_size, void* d_ws, size_t ws_size,
                              hipStream_t stream) {
  const float* feat1 = (const float*)d_in[0];
  const float* feat2 = (const float*)d_in[1];
  const float* flow  = (const float*)d_in[2];
  const float* fW1 = (const float*)d_in[3];  const float* fb1 = (const float*)d_in[4];
  const float* fW2 = (const float*)d_in[5];  const float* fb2 = (const float*)d_in[6];
  const float* fW3 = (const float*)d_in[7];  const float* fb3 = (const float*)d_in[8];
  const float* rW1 = (const float*)d_in[9];  const float* rb1 = (const float*)d_in[10];
  const float* rW2 = (const float*)d_in[11]; const float* rb2 = (const float*)d_in[12];
  const float* rW3 = (const float*)d_in[13]; const float* rb3 = (const float*)d_in[14];
  const float* rW4 = (const float*)d_in[15]; const float* rb4 = (const float*)d_in[16];
  float* out = (float*)d_out;

  // workspace layout (~177 MB; H1/H2/H3 overlay dead feature buffers)
  const size_t SZ32  = (size_t)NPIX * 32  * 2;
  const size_t SZ64  = (size_t)NPIX * 64  * 2;
  const size_t SZ96  = (size_t)NPIX * 96  * 2;
  const size_t SZ128 = (size_t)NPIX * 128 * 2;
  char* ws = (char*)d_ws;
  _Float16* A0  = (_Float16*)(ws);
  _Float16* A1  = (_Float16*)(ws + SZ32);
  _Float16* T0  = (_Float16*)(ws + 2 * SZ32);
  _Float16* T1  = (_Float16*)(ws + 2 * SZ32 + SZ64);
  _Float16* F1  = (_Float16*)(ws + 2 * SZ32 + 2 * SZ64);
  _Float16* F2  = (_Float16*)(ws + 3 * SZ32 + 2 * SZ64);
  _Float16* F2W = (_Float16*)(ws + 4 * SZ32 + 2 * SZ64);
  _Float16* CV  = (_Float16*)(ws + 5 * SZ32 + 2 * SZ64);
  size_t off_wp = 5 * SZ32 + 2 * SZ64 + SZ96;
  _Float16* H1  = (_Float16*)(ws);                       // over A0/A1/T0
  _Float16* H2  = (_Float16*)(ws + SZ128);               // over T1/F1/F2
  _Float16* H3  = (_Float16*)(ws + 5 * SZ32 + 2 * SZ64); // over CV

  auto wpBytes = [](int MT, int KT) { return (size_t)MT * KT * 1024; };
  _Float16* WP1 = (_Float16*)(ws + off_wp);  off_wp += wpBytes(4, 9);
  _Float16* WP2 = (_Float16*)(ws + off_wp);  off_wp += wpBytes(4, 18);
  _Float16* WP3 = (_Float16*)(ws + off_wp);  off_wp += wpBytes(2, 18);
  _Float16* WR1 = (_Float16*)(ws + off_wp);  off_wp += wpBytes(8, 27);
  _Float16* WR2 = (_Float16*)(ws + off_wp);  off_wp += wpBytes(8, 36);
  _Float16* WR3 = (_Float16*)(ws + off_wp);  off_wp += wpBytes(4, 36);
  _Float16* WR4 = (_Float16*)(ws + off_wp);  off_wp += wpBytes(1, 18);

  auto pack = [&](const float* w, _Float16* wp, int CIN, int CINP, int COUT) {
    int KT = CINP * 9 / 32, MT = ((COUT + 15) / 16);
    int threads = MT * KT * 32;
    pack_weight_kernel<<<(threads + 255) / 256, 256, 0, stream>>>(
        w, wp, CIN, CINP, COUT, MT, KT);
  };
  auto conv = [&](const _Float16* act, const _Float16* wp, const float* bias,
                  _Float16* o, float* ofin, const float* fl,
                  int CINP, int COUTP, int relu, int fin) {
    int MT = COUTP / 16;
    int thr = 64 * MT;                       // 2*MT waves per block
    int shmem = 3 * 66 * CINP * 2;
    dim3 grid(NPIX / PSTRIP);                // 3584 strips, exact
    switch (CINP) {
      case 32:
        conv3x3_wmma_lds_kernel<32><<<grid, thr, shmem, stream>>>(
            act, wp, bias, o, ofin, fl, MT, COUTP, relu, fin); break;
      case 64:
        conv3x3_wmma_lds_kernel<64><<<grid, thr, shmem, stream>>>(
            act, wp, bias, o, ofin, fl, MT, COUTP, relu, fin); break;
      case 96:
        conv3x3_wmma_lds_kernel<96><<<grid, thr, shmem, stream>>>(
            act, wp, bias, o, ofin, fl, MT, COUTP, relu, fin); break;
      case 128:
        conv3x3_wmma_lds_kernel<128><<<grid, thr, shmem, stream>>>(
            act, wp, bias, o, ofin, fl, MT, COUTP, relu, fin); break;
    }
  };

  const int PB = (NPIX + 255) / 256;

  // weight packing (re-done every call: deterministic, graph-safe)
  pack(fW1, WP1, 3,  32, 64);
  pack(fW2, WP2, 64, 64, 64);
  pack(fW3, WP3, 64, 64, 32);
  pack(rW1, WR1, 83, 96, 128);
  pack(rW2, WR2, 128, 128, 128);
  pack(rW3, WR3, 128, 128, 64);
  pack(rW4, WR4, 64, 64, 2);

  // inputs -> padded channel-last f16
  pad_input_kernel<<<PB, 256, 0, stream>>>(feat1, A0);
  pad_input_kernel<<<PB, 256, 0, stream>>>(feat2, A1);

  // feature extraction (both images)
  conv(A0, WP1, fb1, T0, nullptr, nullptr, 32, 64, 1, 0);
  conv(T0, WP2, fb2, T1, nullptr, nullptr, 64, 64, 1, 0);
  conv(T1, WP3, fb3, F1, nullptr, nullptr, 64, 32, 1, 0);
  conv(A1, WP1, fb1, T0, nullptr, nullptr, 32, 64, 1, 0);
  conv(T0, WP2, fb2, T1, nullptr, nullptr, 64, 64, 1, 0);
  conv(T1, WP3, fb3, F2, nullptr, nullptr, 64, 32, 1, 0);

  // warp + cost volume (+flow, zero-pad to 96 channels)
  warp_kernel<<<PB, 256, 0, stream>>>(F2, flow, F2W);
  costvol_kernel<<<PB, 256, 0, stream>>>(F1, F2W, flow, CV);

  // refinement head
  conv(CV, WR1, rb1, H1, nullptr, nullptr, 96, 128, 1, 0);
  conv(H1, WR2, rb2, H2, nullptr, nullptr, 128, 128, 1, 0);
  conv(H2, WR3, rb3, H3, nullptr, nullptr, 128, 64, 1, 0);
  conv(H3, WR4, rb4, nullptr, out, flow, 64, 16, 0, 1);   // fused +flow head
}
